// SMCN_87978110091684
// MI455X (gfx1250) — matrix operand order repaired and checked by reading, not compiled, not run on hardware
//
#include <hip/hip_runtime.h>

// ---------------- problem constants (from reference) ----------------
constexpr int kT   = 256;
constexpr int kBS  = 16;
constexpr int kIN  = 64;
constexpr int kLAT = 256;
constexpr int kOUT = 32;
constexpr int kNP  = 128;
constexpr int kTHREADS = 512;   // 16 wave32s

// ---------------- types ----------------
typedef __bf16 bf16_t;
typedef bf16_t v16bf __attribute__((ext_vector_type(16)));
typedef bf16_t v8bf  __attribute__((ext_vector_type(8)));
typedef float  v8f   __attribute__((ext_vector_type(8)));
typedef float  v4f   __attribute__((ext_vector_type(4)));

// ---------------- helpers ----------------
__device__ __forceinline__ unsigned hash_u32(unsigned x) {
  x ^= x >> 17; x *= 0xed5ad4bbu;
  x ^= x >> 11; x *= 0xac4c1b51u;
  x ^= x >> 15; x *= 0x31848babu;
  x ^= x >> 14; return x;
}
__device__ __forceinline__ float u01(unsigned x) {
  return (hash_u32(x) >> 8) * (1.0f / 16777216.0f);
}
__device__ __forceinline__ float gaussn(unsigned s) {
  float a = fmaxf(u01(s), 1e-7f);
  float b = u01(s ^ 0x9e3779b9u);
  return __fsqrt_rn(-2.0f * __logf(a)) * __cosf(6.28318530718f * b);
}
__device__ __forceinline__ float sigmoidf(float x) {
  return 1.0f / (1.0f + __expf(-x));
}

// Optimization barrier: an opaque zero offset. Adding it to a pointer keeps
// the pointer's provenance/address-space (loads stay global_load_b128 with a
// scalar base, NOT flat/SYS), but the optimizer cannot prove the resulting
// address equals any previous one, so weight-fragment loads cannot be
// hoisted/CSE'd across loop iterations (which previously caused either
// 192-VGPR caching + scratch spills, or serialized volatile flats).
__device__ __forceinline__ size_t opaque_zero() {
  size_t z = 0;
  asm volatile("" : "+s"(z));
  return z;
}

__device__ __forceinline__ v8f wmma_bf16(v16bf a, v16bf b, v8f c) {
  // D = A(16x32 bf16) x B(32x16 bf16) + C(16x16 f32)
  return __builtin_amdgcn_wmma_f32_16x16x32_bf16(
      false, a, false, b, (short)0, c, false, false);
}

// A fragment (16x32, bf16) from an fp32 row-major matrix, converting on the fly.
// ISA layout: lane L (m = L&15, s = L>>4): halfs 0..7 = K s*8+0..7, halfs 8..15 = K 16+s*8+0..7
__device__ __forceinline__ v16bf load_a_from_f32(const float* __restrict__ A, int lda,
                                                 int mt, int kt, int lane) {
  const int m  = mt * 16 + (lane & 15);
  const int k0 = kt * 32 + ((lane >> 4) << 3);
  const v4f* p = reinterpret_cast<const v4f*>(A + (size_t)m * lda + k0);
  v4f x0 = p[0], x1 = p[1], x2 = p[4], x3 = p[5];
  v16bf a;
#pragma unroll
  for (int i = 0; i < 4; ++i) {
    a[i]      = (bf16_t)x0[i];
    a[4 + i]  = (bf16_t)x1[i];
    a[8 + i]  = (bf16_t)x2[i];
    a[12 + i] = (bf16_t)x3[i];
  }
  return a;
}

// A fragment (16x32) from a bf16 row-major matrix.
__device__ __forceinline__ v16bf load_a_bf16rm(const bf16_t* __restrict__ A, int lda,
                                               int mt, int kt, int lane) {
  const int m  = mt * 16 + (lane & 15);
  const int k0 = kt * 32 + ((lane >> 4) << 3);
  const v8bf* p = reinterpret_cast<const v8bf*>(A + (size_t)m * lda + k0);
  v8bf b0 = p[0], b1 = p[2];  // +16 halfs
  v16bf r;
#pragma unroll
  for (int i = 0; i < 8; ++i) { r[i] = b0[i]; r[8 + i] = b1[i]; }
  return r;
}

// B fragment (32x16, bf16). B = W^T where W is (N,K) row-major; lane half s
// takes K s*8+0..7 and 16+s*8+0..7 from W's row n (matches the A layout).
__device__ __forceinline__ v16bf load_b_bf16rm(const bf16_t* __restrict__ W, int ldw,
                                               int nrow0, int kt, int lane) {
  const int n  = nrow0 + (lane & 15);
  const int k0 = kt * 32 + ((lane >> 4) << 3);
  const v8bf* p = reinterpret_cast<const v8bf*>(W + (size_t)n * ldw + k0);
  v8bf b0 = p[0], b1 = p[2];
  v16bf r;
#pragma unroll
  for (int i = 0; i < 8; ++i) { r[i] = b0[i]; r[8 + i] = b1[i]; }
  return r;
}

// ---------------- kernel 1: weight conversion + state init ----------------
__global__ void smcn_init_kernel(const float* __restrict__ g_whh,
                                 const float* __restrict__ f_w1,
                                 const float* __restrict__ f_w2,
                                 unsigned short* __restrict__ whh16,
                                 unsigned short* __restrict__ w116,
                                 unsigned short* __restrict__ w216,
                                 float* __restrict__ h, float* __restrict__ hg) {
  const int n_whh = 3 * kLAT * kLAT;
  const int n_w1  = kLAT * kLAT;
  const int n_w2  = kOUT * kLAT;
  const int n_h   = kBS * kNP * kLAT;
  bf16_t* Wa = reinterpret_cast<bf16_t*>(whh16);
  bf16_t* Wb = reinterpret_cast<bf16_t*>(w116);
  bf16_t* Wc = reinterpret_cast<bf16_t*>(w216);
  for (int i = blockIdx.x * blockDim.x + threadIdx.x; i < n_h;
       i += gridDim.x * blockDim.x) {
    if (i < n_whh) Wa[i] = (bf16_t)g_whh[i];
    if (i < n_w1)  Wb[i] = (bf16_t)f_w1[i];
    if (i < n_w2)  Wc[i] = (bf16_t)f_w2[i];
    h[i]  = 0.0f;
    hg[i] = 0.0f;
  }
}

// ---------------- kernel 2: 3-layer GRU input model (tiny, scalar) ----------------
__global__ __launch_bounds__(64) void smcn_gru3_kernel(
    const float* __restrict__ u,
    const float* wi0, const float* wh0, const float* bi0, const float* bh0,
    const float* wi1, const float* wh1, const float* bi1, const float* bh1,
    const float* wi2, const float* wh2, const float* bi2, const float* bh2,
    float* __restrict__ up) {
  const int b = blockIdx.x;
  const int j = threadIdx.x;
  __shared__ float sx[kIN];
  __shared__ float sh[3][kIN];
  sh[0][j] = 0.f; sh[1][j] = 0.f; sh[2][j] = 0.f;
  const float* Wi[3] = {wi0, wi1, wi2};
  const float* Wh[3] = {wh0, wh1, wh2};
  const float* Bi[3] = {bi0, bi1, bi2};
  const float* Bh[3] = {bh0, bh1, bh2};
  __syncthreads();
  for (int t = 0; t < kT; ++t) {
    sx[j] = u[((size_t)t * kBS + b) * kIN + j];
    __syncthreads();
    for (int l = 0; l < 3; ++l) {
      float ir = Bi[l][j], iz = Bi[l][kIN + j], in_ = Bi[l][2 * kIN + j];
      float hr = Bh[l][j], hz = Bh[l][kIN + j], hn  = Bh[l][2 * kIN + j];
      const float* wir = Wi[l] + (size_t)j * kIN;
      const float* wiz = Wi[l] + (size_t)(kIN + j) * kIN;
      const float* win = Wi[l] + (size_t)(2 * kIN + j) * kIN;
      const float* whr = Wh[l] + (size_t)j * kIN;
      const float* whz = Wh[l] + (size_t)(kIN + j) * kIN;
      const float* whn = Wh[l] + (size_t)(2 * kIN + j) * kIN;
      for (int k2 = 0; k2 < kIN; ++k2) {
        const float xv = sx[k2], hv = sh[l][k2];
        ir += wir[k2] * xv; iz += wiz[k2] * xv; in_ += win[k2] * xv;
        hr += whr[k2] * hv; hz += whz[k2] * hv; hn  += whn[k2] * hv;
      }
      const float r = sigmoidf(ir + hr);
      const float z = sigmoidf(iz + hz);
      const float n = tanhf(in_ + r * hn);
      const float hnew = (1.f - z) * n + z * sh[l][j];
      __syncthreads();
      sh[l][j] = hnew;
      sx[j]    = hnew;
      __syncthreads();
    }
    up[((size_t)t * kBS + b) * kIN + j] = sx[j];
    __syncthreads();
  }
}

// ---------------- kernel 3: persistent particle filter, WMMA bf16 ----------------
// 512 threads = 16 wave32s. Wave wv owns latent j-tile jt = wv (16 cols)
// across the 3 GRU gates. B fragments stream from L2 each mt iteration via an
// opaque zero offset -> normal cached global_load_b128, no hoist, no spill.
__global__ __launch_bounds__(kTHREADS) void smcn_pf_kernel(
    const float* __restrict__ up, const float* __restrict__ y,
    const float* __restrict__ g_wih, const float* __restrict__ g_bih,
    const float* __restrict__ g_bhh,
    const unsigned short* __restrict__ whh16_,
    const unsigned short* __restrict__ w116_,
    const unsigned short* __restrict__ w216_,
    const float* __restrict__ f_b1, const float* __restrict__ f_b2,
    const float* __restrict__ logvar_x, const float* __restrict__ logvar_y,
    float* __restrict__ hglob, float* __restrict__ hgglob,
    unsigned short* __restrict__ hid16_, float* __restrict__ out) {
  const int b    = blockIdx.x;
  const int tid  = threadIdx.x;
  const int lane = tid & 31;
  const int wv   = tid >> 5;     // 16 waves
  const int hi   = lane >> 4;
  const int jl   = lane & 15;

  const bf16_t* Whh = reinterpret_cast<const bf16_t*>(whh16_);
  const bf16_t* W1  = reinterpret_cast<const bf16_t*>(w116_);
  const bf16_t* W2  = reinterpret_cast<const bf16_t*>(w216_);
  bf16_t* hidB = reinterpret_cast<bf16_t*>(hid16_) + (size_t)b * kNP * kLAT;
  float*  hB   = hglob  + (size_t)b * kNP * kLAT;
  float*  hgB  = hgglob + (size_t)b * kNP * kLAT;

  __shared__ float s_gi[3 * kLAT];       // 3 KB
  __shared__ float s_yhat[kNP * kOUT];   // 16 KB
  __shared__ float s_logp[kNP];
  __shared__ float s_cum[kNP];
  __shared__ int   s_idx[kNP];
  __shared__ float s_tot;

  const v8f vzero = {};

  for (int t = 0; t < kT; ++t) {
    // ---------- resampling (t >= 1): categorical on diag-MVN logp ----------
    if (t > 0) {
      const float* yprev = y + ((size_t)(t - 1) * kBS + b) * kOUT;
      if (tid < kNP) {
        float acc = 0.f;
        for (int o = 0; o < kOUT; ++o) {
          const float d = s_yhat[tid * kOUT + o] - yprev[o];
          acc += d * d * __expf(-logvar_y[o]);
        }
        s_logp[tid] = -0.5f * acc;   // constants cancel in softmax
      }
      __syncthreads();
      if (tid == 0) {
        float mx = s_logp[0];
        for (int m = 1; m < kNP; ++m) mx = fmaxf(mx, s_logp[m]);
        float csum = 0.f;
        for (int m = 0; m < kNP; ++m) { csum += __expf(s_logp[m] - mx); s_cum[m] = csum; }
        s_tot = csum;
      }
      __syncthreads();
      if (tid < kNP) {
        const float ur = u01((((unsigned)t * kBS + b) * kNP + tid) | 0x40000000u) * s_tot;
        int sel = kNP - 1;
        for (int m = 0; m < kNP; ++m) { if (ur < s_cum[m]) { sel = m; break; } }
        s_idx[tid] = sel;
      }
      __syncthreads();
      // gather resampled particles into hg
      for (int e = tid; e < kNP * kLAT; e += kTHREADS) {
        const int m = e >> 8;              // kLAT == 256
        const int j = e & (kLAT - 1);
        hgB[e] = hB[(size_t)s_idx[m] * kLAT + j];
      }
      __threadfence();
    }
    __syncthreads();

    // ---------- gi = up[t] @ g_wih^T + g_bih (shared across particles) ----------
    const float* ut = up + ((size_t)t * kBS + b) * kIN;
    for (int r0 = tid; r0 < 3 * kLAT; r0 += kTHREADS) {
      const float* wr = g_wih + (size_t)r0 * kIN;
      float acc = g_bih[r0];
#pragma unroll 8
      for (int k2 = 0; k2 < kIN; ++k2) acc += ut[k2] * wr[k2];
      s_gi[r0] = acc;
    }
    // keep recurrent weights warm in cache (global_prefetch_b8)
    __builtin_prefetch(Whh + (size_t)(wv * 16 + jl) * kLAT, 0, 1);
    __syncthreads();

    // ---------- particle GRU cell: gh = hg @ Whh^T, gates fused in epilogue ----------
    const int j = wv * 16 + jl;            // this wave's latent column
    const float gir = s_gi[j], giz = s_gi[kLAT + j], gin = s_gi[2 * kLAT + j];
    const float bhr = g_bhh[j], bhz = g_bhh[kLAT + j], bhn = g_bhh[2 * kLAT + j];
    const float stdx = __expf(0.5f * logvar_x[j]);

    for (int mt = 0; mt < 8; ++mt) {
      const bf16_t* Wl = Whh + opaque_zero();  // fresh opaque base per mt iteration
      v8f cr = vzero, cz = vzero, cn = vzero;
#pragma unroll
      for (int kt = 0; kt < 8; ++kt) {
        const v16bf a = load_a_from_f32(hgB, kLAT, mt, kt, lane);
        cr = wmma_bf16(a, load_b_bf16rm(Wl, kLAT, 0 * kLAT + wv * 16, kt, lane), cr);
        cz = wmma_bf16(a, load_b_bf16rm(Wl, kLAT, 1 * kLAT + wv * 16, kt, lane), cz);
        cn = wmma_bf16(a, load_b_bf16rm(Wl, kLAT, 2 * kLAT + wv * 16, kt, lane), cn);
      }
#pragma unroll
      for (int p = 0; p < 8; ++p) {
        const int m = mt * 16 + p + 8 * hi;
        const float ho = hgB[(size_t)m * kLAT + j];
        const float r  = sigmoidf(gir + cr[p] + bhr);
        const float z  = sigmoidf(giz + cz[p] + bhz);
        const float nn = tanhf(gin + r * (cn[p] + bhn));
        float hv = (1.f - z) * nn + z * ho;
        hv += stdx * gaussn((((unsigned)t * kBS + b) * kNP + m) * kLAT + j);
        hB[(size_t)m * kLAT + j] = hv;
      }
    }
    __threadfence();
    __syncthreads();

    // ---------- FFN layer 1: hid = relu(h @ W1^T + b1); wave owns n-tile wv ----------
    {
      const int n = wv * 16 + jl;
      const float bias = f_b1[n];
      for (int mt = 0; mt < 8; ++mt) {
        const bf16_t* W1l = W1 + opaque_zero();
        v8f c = vzero;
#pragma unroll
        for (int kt = 0; kt < 8; ++kt) {
          const v16bf a = load_a_from_f32(hB, kLAT, mt, kt, lane);
          c = wmma_bf16(a, load_b_bf16rm(W1l, kLAT, wv * 16, kt, lane), c);
        }
#pragma unroll
        for (int p = 0; p < 8; ++p) {
          const int m = mt * 16 + p + 8 * hi;
          hidB[(size_t)m * kLAT + n] = (bf16_t)fmaxf(c[p] + bias, 0.f);
        }
      }
    }
    __threadfence();
    __syncthreads();

    // ---------- FFN layer 2: y_hat = hid @ W2^T + b2; wave owns (mt, n-half) ----------
    {
      const int mt2 = wv & 7;
      const int nn2 = wv >> 3;             // 0 or 1 -> output cols 0..15 / 16..31
      const bf16_t* W2l = W2 + opaque_zero();  // prevent hoist out of the t loop
      v8f c = vzero;
#pragma unroll
      for (int kt = 0; kt < 8; ++kt) {
        const v16bf a = load_a_bf16rm(hidB, kLAT, mt2, kt, lane);
        c = wmma_bf16(a, load_b_bf16rm(W2l, kLAT, nn2 * 16, kt, lane), c);
      }
      float* outT = out + (((size_t)t * kBS + b) * kNP) * kOUT;
      const int o = nn2 * 16 + jl;
      const float bias = f_b2[o];
#pragma unroll
      for (int p = 0; p < 8; ++p) {
        const int m = mt2 * 16 + p + 8 * hi;
        const float v = c[p] + bias;
        s_yhat[m * kOUT + o] = v;
        outT[(size_t)m * kOUT + o] = v;
      }
    }
    __syncthreads();   // y_hat ready for next step's resampling
  }
}

// ---------------- host ----------------
extern "C" void kernel_launch(void* const* d_in, const int* in_sizes, int n_in,
                              void* d_out, int out_size, void* d_ws, size_t ws_size,
                              hipStream_t stream) {
  (void)in_sizes; (void)n_in; (void)out_size; (void)ws_size;

  const float* u        = (const float*)d_in[0];
  const float* y        = (const float*)d_in[1];
  const float* g_wih    = (const float*)d_in[2];
  const float* g_whh    = (const float*)d_in[3];
  const float* g_bih    = (const float*)d_in[4];
  const float* g_bhh    = (const float*)d_in[5];
  const float* f_w1     = (const float*)d_in[6];
  const float* f_b1     = (const float*)d_in[7];
  const float* f_w2     = (const float*)d_in[8];
  const float* f_b2     = (const float*)d_in[9];
  const float* logvar_x = (const float*)d_in[10];
  const float* logvar_y = (const float*)d_in[11];
  const float* wi0 = (const float*)d_in[12];
  const float* wh0 = (const float*)d_in[13];
  const float* bi0 = (const float*)d_in[14];
  const float* bh0 = (const float*)d_in[15];
  const float* wi1 = (const float*)d_in[16];
  const float* wh1 = (const float*)d_in[17];
  const float* bi1 = (const float*)d_in[18];
  const float* bh1 = (const float*)d_in[19];
  const float* wi2 = (const float*)d_in[20];
  const float* wh2 = (const float*)d_in[21];
  const float* bi2 = (const float*)d_in[22];
  const float* bh2 = (const float*)d_in[23];

  char* ws = (char*)d_ws;
  size_t off = 0;
  float* up_ws = (float*)(ws + off);            off += (size_t)kT * kBS * kIN * 4;
  unsigned short* whh16 = (unsigned short*)(ws + off); off += (size_t)3 * kLAT * kLAT * 2;
  unsigned short* w116  = (unsigned short*)(ws + off); off += (size_t)kLAT * kLAT * 2;
  unsigned short* w216  = (unsigned short*)(ws + off); off += (size_t)kOUT * kLAT * 2;
  float* hbuf  = (float*)(ws + off);            off += (size_t)kBS * kNP * kLAT * 4;
  float* hgbuf = (float*)(ws + off);            off += (size_t)kBS * kNP * kLAT * 4;
  unsigned short* hid16 = (unsigned short*)(ws + off); off += (size_t)kBS * kNP * kLAT * 2;

  float* out = (float*)d_out;

  smcn_init_kernel<<<2048, 256, 0, stream>>>(g_whh, f_w1, f_w2, whh16, w116, w216,
                                             hbuf, hgbuf);
  smcn_gru3_kernel<<<kBS, 64, 0, stream>>>(u, wi0, wh0, bi0, bh0,
                                           wi1, wh1, bi1, bh1,
                                           wi2, wh2, bi2, bh2, up_ws);
  smcn_pf_kernel<<<kBS, kTHREADS, 0, stream>>>(up_ws, y, g_wih, g_bih, g_bhh,
                                               whh16, w116, w216, f_b1, f_b2,
                                               logvar_x, logvar_y,
                                               hbuf, hgbuf, hid16, out);
}
